// VectorQuantizer_65781719105537
// MI455X (gfx1250) — matrix operand order repaired and checked by reading, not compiled, not run on hardware
//
#include <hip/hip_runtime.h>

// ---------------------------------------------------------------------------
// VQ-VAE vector quantizer for MI455X (gfx1250, wave32, WMMA).
//
// score[n,c] = e2[c] - 2 * (x[n] . emb[c])   (x2/D dropped: monotone per row)
// GEMM on the bf16 matrix pipe (v_wmma_f32_16x16x32_bf16) with an hi/lo bf16
// split of x for ~fp32 accuracy: x.e ~= hi.hi + hi.lo + lo.hi.
// Each wave processes TWO 16-row tiles per iteration so every B operand
// loaded from LDS feeds 2 WMMAs (12 WMMAs / 4KB of LDS traffic).
// __launch_bounds__(256,2): keep ~A/acc/key state in VGPRs (no scratch
// spills); LDS (130KB/WG) already caps us at <=2 WGs per WGP.
// Per-lane argmin: plain float+idx (strict <, codes ascend per lane ->
// first-occurrence). Cross-lane: order-preserving u64 key (f32order<<9|code),
// branchless min over __shfl_xor, exact jnp.argmin tie-break.
// Loss = 1.25 * mean((quantized - x)^2); out = quantized values, NCHW.
// ---------------------------------------------------------------------------

typedef __attribute__((ext_vector_type(16))) __bf16 v16bf;
typedef __attribute__((ext_vector_type(8)))  float  v8f;

#define CODEBOOK   512
#define EMBD       64
#define BATCH      32
#define HW         4096
#define NROWS      (BATCH * HW)        // 131072
#define NTILES     (NROWS / 16)        // 8192 row tiles
#define NPAIRS     (NTILES / 2)        // 4096 dual-tile units
#define CTILES     (CODEBOOK / 16)     // 32 code tiles
#define OUT_ELEMS  (NROWS * EMBD)      // 8388608
#define IDX_OFF    (1 + OUT_ELEMS)

#define WGS           128
#define TPB           256
#define WAVES_PER_WG  (TPB / 32)
#define TOTAL_WAVES   (WGS * WAVES_PER_WG)   // 1024 (== partial slots)

// LDS: [0,64K) bf16 hi of emb (transposed, pair-interleaved)
//      [64K,128K) bf16 lo, same layout ; [128K,130K) e2[512] f32
// word index for (code k, dim d):  (d>>1)*512 + k, halves {d even, d odd}
// -> matches WMMA B(32x16) layout: lane c, VGPR v holds K={2v,2v+1}+8*half (+8 more for v>=4).
#define LDS_BYTES  (65536 * 2 + 2048)

__device__ __forceinline__ unsigned short f32_to_bf16_rne(float x) {
    unsigned u = __builtin_bit_cast(unsigned, x);
    unsigned r = u + 0x7FFFu + ((u >> 16) & 1u);
    return (unsigned short)(r >> 16);
}
__device__ __forceinline__ float bf16_bits_to_f32(unsigned short h) {
    unsigned u = ((unsigned)h) << 16;
    return __builtin_bit_cast(float, u);
}
// monotone float -> u32 (ascending order preserved as unsigned compare)
__device__ __forceinline__ unsigned f32_order(float f) {
    unsigned u = __builtin_bit_cast(unsigned, f);
    return (u & 0x80000000u) ? ~u : (u | 0x80000000u);
}

extern "C" __global__ void __launch_bounds__(TPB, 2)
vq_main(const float* __restrict__ inp,
        const float* __restrict__ emb,
        float* __restrict__ outbuf,
        float* __restrict__ partials)
{
    extern __shared__ char smem[];
    unsigned*       bh32 = (unsigned*)smem;
    unsigned*       bl32 = (unsigned*)(smem + 65536);
    unsigned short* bh16 = (unsigned short*)smem;
    unsigned short* bl16 = (unsigned short*)(smem + 65536);
    float*          e2   = (float*)(smem + 131072);

    const int tid = threadIdx.x;

    // ---- stage embedding: transposed pair-interleaved bf16 hi/lo + e2 ----
    for (int j = tid; j < CODEBOOK * EMBD; j += TPB) {
        int k = j >> 6, d = j & 63;
        float v = emb[j];
        unsigned short h  = f32_to_bf16_rne(v);
        unsigned short lo = f32_to_bf16_rne(v - bf16_bits_to_f32(h));
        int us = (((d >> 1) * CODEBOOK + k) << 1) + (d & 1);
        bh16[us] = h;
        bl16[us] = lo;
    }
    for (int k = tid; k < CODEBOOK; k += TPB) {
        const float* er = emb + k * EMBD;
        float s = 0.f;
        #pragma unroll
        for (int d = 0; d < EMBD; ++d) s += er[d] * er[d];
        e2[k] = s;
    }
    __syncthreads();

    const int lane = tid & 31;
    const int m    = lane & 15;     // row within tile / column within code tile
    const int g    = lane >> 4;     // half-wave group
    const int wid  = blockIdx.x * WAVES_PER_WG + (tid >> 5);

    float lsum = 0.f;

    for (int pw = wid; pw < NPAIRS; pw += TOTAL_WAVES) {
        const int n0  = pw << 5;               // 32 consecutive rows, same image b
        const int b   = n0 >> 12;
        const int hw0 = n0 & 4095;
        const float* xb = inp + (size_t)b * (EMBD * HW) + hw0 + m;

        // ---- build A operands for both 16-row tiles (hi/lo split) ----
        // ISA 16-bit A layout: lane m, group g; elem e -> VGPR v=e>>1,
        // K = base(v) + (e&1) + 8g,  base = {0,2,4,6,16,18,20,22}
        v16bf a_hi[2][2], a_lo[2][2];          // [tile][kblock]
        #pragma unroll
        for (int u = 0; u < 2; ++u) {
            #pragma unroll
            for (int kb = 0; kb < 2; ++kb) {
                union { unsigned short us[16]; v16bf v; } uh, ul;
                #pragma unroll
                for (int e = 0; e < 16; ++e) {
                    int vv   = e >> 1;
                    int base = (vv < 4) ? (2 * vv) : (8 + 2 * vv);
                    int d    = base + (e & 1) + 8 * g + 32 * kb;
                    float x  = xb[(size_t)d * HW + 16 * u];     // coalesced
                    unsigned short h = f32_to_bf16_rne(x);
                    uh.us[e] = h;
                    ul.us[e] = f32_to_bf16_rne(x - bf16_bits_to_f32(h));
                }
                a_hi[u][kb] = uh.v;
                a_lo[u][kb] = ul.v;
            }
        }

        float bestv[2][8];
        int   besti[2][8];
        #pragma unroll
        for (int u = 0; u < 2; ++u)
            #pragma unroll
            for (int r = 0; r < 8; ++r) { bestv[u][r] = 3.4e38f; besti[u][r] = 0; }

        // ---- 32 code tiles: each B operand pair feeds both row tiles ----
        for (int ct = 0; ct < CTILES; ++ct) {
            const int c = ct * 16 + m;
            v8f acc[2] = {{}, {}};
            #pragma unroll
            for (int kb = 0; kb < 2; ++kb) {
                union { unsigned w[8]; v16bf v; } Bh, Bl;
                #pragma unroll
                for (int pe = 0; pe < 8; ++pe) {
                    int p = ((pe < 4) ? pe : pe + 4) + 4 * g + 16 * kb;      // K-pair index
                    Bh.w[pe] = bh32[p * CODEBOOK + c];
                    Bl.w[pe] = bl32[p * CODEBOOK + c];
                }
                #pragma unroll
                for (int u = 0; u < 2; ++u) {          // 12 WMMAs / ct
                    acc[u] = __builtin_amdgcn_wmma_f32_16x16x32_bf16(
                        false, a_hi[u][kb], false, Bh.v, (short)0, acc[u], false, false);
                    acc[u] = __builtin_amdgcn_wmma_f32_16x16x32_bf16(
                        false, a_hi[u][kb], false, Bl.v, (short)0, acc[u], false, false);
                    acc[u] = __builtin_amdgcn_wmma_f32_16x16x32_bf16(
                        false, a_lo[u][kb], false, Bh.v, (short)0, acc[u], false, false);
                }
            }
            const float e2c = e2[c];
            #pragma unroll
            for (int u = 0; u < 2; ++u) {
                #pragma unroll
                for (int r = 0; r < 8; ++r) {          // C/D layout: VGPR r, M=r+8g
                    float s = fmaf(-2.f, acc[u][r], e2c);
                    // codes ascend per lane -> strict < keeps first occurrence
                    if (s < bestv[u][r]) { bestv[u][r] = s; besti[u][r] = c; }
                }
            }
        }

        // ---- branchless argmin across the 16 lanes of each half ----
        // pack once: monotone(f32) in high bits, 9-bit code in low bits
        unsigned long long key[2][8];
        #pragma unroll
        for (int u = 0; u < 2; ++u)
            #pragma unroll
            for (int r = 0; r < 8; ++r)
                key[u][r] = ((unsigned long long)f32_order(bestv[u][r]) << 9)
                          | (unsigned)besti[u][r];
        #pragma unroll
        for (int off = 1; off < 16; off <<= 1) {
            #pragma unroll
            for (int u = 0; u < 2; ++u) {
                #pragma unroll
                for (int r = 0; r < 8; ++r) {
                    unsigned long long ok = __shfl_xor(key[u][r], off, 32);
                    key[u][r] = (ok < key[u][r]) ? ok : key[u][r];
                }
            }
        }

        // ---- per-tile outputs ----
        #pragma unroll
        for (int u = 0; u < 2; ++u) {
            // row m's winner: register (m&7) of lanes with half == (m>>3)
            int irow = 0;
            const int srcl = (m & 8) ? 16 : 0;
            #pragma unroll
            for (int r = 0; r < 8; ++r) {
                int code = (int)(key[u][r] & 0x1FFull);
                int tmp  = __shfl(code, srcl, 32);
                if ((m & 7) == r) irow = tmp;
            }

            const float* erow  = emb + (size_t)irow * EMBD;   // L2-hot table
            const size_t obase = (size_t)b * (EMBD * HW) + (hw0 + 16 * u) + m;
            const int dlo = g * 32;
            #pragma unroll 8
            for (int d = dlo; d < dlo + 32; ++d) {
                float q  = erow[d];
                float x  = inp[obase + (size_t)d * HW];
                float df = q - x;
                lsum += df * df;
                outbuf[1 + obase + (size_t)d * HW] = q;       // coalesced per d
            }
            if (g == 0) outbuf[IDX_OFF + (n0 + 16 * u) + m] = (float)irow;
        }
    }

    // deterministic loss: one fixed slot per wave, no float atomics
    #pragma unroll
    for (int off = 1; off < 32; off <<= 1) lsum += __shfl_xor(lsum, off, 32);
    if (lane == 0) partials[wid] = lsum;
}

extern "C" __global__ void vq_reduce(const float* __restrict__ partials,
                                     float* __restrict__ outbuf)
{
    __shared__ float sm[256];
    float s = 0.f;
    for (int i = threadIdx.x; i < TOTAL_WAVES; i += 256) s += partials[i];
    sm[threadIdx.x] = s;
    __syncthreads();
    for (int off = 128; off > 0; off >>= 1) {
        if ((int)threadIdx.x < off) sm[threadIdx.x] += sm[threadIdx.x + off];
        __syncthreads();
    }
    if (threadIdx.x == 0)
        outbuf[0] = sm[0] * (1.25f / (float)OUT_ELEMS);   // q_loss + 0.25*e_loss
}

extern "C" void kernel_launch(void* const* d_in, const int* in_sizes, int n_in,
                              void* d_out, int out_size, void* d_ws, size_t ws_size,
                              hipStream_t stream)
{
    const float* inp = (const float*)d_in[0];   // [32,64,64,64] f32 NCHW
    const float* emb = (const float*)d_in[1];   // [512,64] f32
    float* out      = (float*)d_out;            // [1 + 8388608 + 131072] f32
    float* partials = (float*)d_ws;             // 1024 f32 wave partials

    vq_main<<<WGS, TPB, LDS_BYTES, stream>>>(inp, emb, out, partials);
    vq_reduce<<<1, 256, 0, stream>>>(partials, out);
}
